// MeshEncoderDecoder_59098749993125
// MI455X (gfx1250) — matrix operand order repaired and checked by reading, not compiled, not run on hardware
//
#include <hip/hip_runtime.h>
#include <math.h>

#define E_FINE 22000
#define E_COARSE 3000
#define NBATCH 4

typedef __attribute__((ext_vector_type(16))) _Float16 v16h;
typedef __attribute__((ext_vector_type(8)))  float    v8f;

// ---------------------------------------------------------------------------
// Weight prep: f32 [O][CinK] -> f16 [Opad][Kpad], zero padded.
// ---------------------------------------------------------------------------
__global__ __launch_bounds__(256) void wprep_kernel(
    const float* __restrict__ W, _Float16* __restrict__ dst,
    int O, int CinK, int Kpad, int Opad) {
  int idx = blockIdx.x * blockDim.x + threadIdx.x;
  if (idx >= Opad * Kpad) return;
  int o = idx / Kpad, k = idx - o * Kpad;
  float v = (o < O && k < CinK) ? W[o * CinK + k] : 0.0f;
  dst[idx] = (_Float16)v;
}

// ---------------------------------------------------------------------------
// WMMA GEMM over 32-edge tiles.  out[b,o,e] = bias[o] + sum_k Wk[o,k]*F[k,e]
//   gidx == null : F[c,e]      = src[b,c,e]                     (conv1d, K=Cin)
//   gidx != null : F[c*5+q,e]  = mesh features of src via gemm   (K=Cin*5)
// Block: (Opad/16) waves; each wave owns one 16(outch) x 32(edge) tile as two
// 16x16 accumulators sharing the A (weight) fragment -> 2x A reuse per load.
// LDS: two banks of f16 features [Kpad][16] (one per 16-edge sub-tile), each
// K-pair interleaved so one B operand's dwords sit at a uniform 64B stride
// (clean ds_load_2addr pairs straight into the WMMA source registers).
// ---------------------------------------------------------------------------
__global__ __launch_bounds__(320) void wmma_gemm_kernel(
    const float* __restrict__ src, long srcBS, int Cin,
    const int* __restrict__ gidx,
    const _Float16* __restrict__ Wk, const float* __restrict__ bias,
    float* __restrict__ dst, long dstBS, int O, int Kpad, int E) {
  extern __shared__ _Float16 lF[];   // 2 banks * Kpad*16 halves
  const int bank = Kpad * 16;        // halves per sub-tile bank
  const int b  = blockIdx.y;
  const int e0 = blockIdx.x * 32;
  const int tid = threadIdx.x, nth = blockDim.x;
  const float* sb = src + (long)b * srcBS;
  const int Kact = gidx ? Cin * 5 : Cin;

  if (gidx) {
    const int* gb = gidx + ((long)b * E + e0) * 4;
    for (int idx = tid; idx < Cin * 32; idx += nth) {
      int c = idx >> 5, j = idx & 31;
      int e = e0 + j;
      float v[5] = {0.f, 0.f, 0.f, 0.f, 0.f};
      if (e < E) {
        const float* row = sb + (long)c * E;
        float x0 = row[e];
        int g0 = gb[j*4+0], g1 = gb[j*4+1], g2 = gb[j*4+2], g3 = gb[j*4+3];
        float f1 = row[g0], f2 = row[g1], f3 = row[g2], f4 = row[g3];
        v[0] = x0; v[1] = f1 + f3; v[2] = f2 + f4;
        v[3] = fabsf(f1 - f3); v[4] = fabsf(f2 - f4);
      }
      _Float16* bk = lF + (j >> 4) * bank;
      int jj = j & 15;
      int kb = c * 5;
#pragma unroll
      for (int q = 0; q < 5; q++) {
        int k = kb + q;
        bk[(k >> 1) * 32 + jj * 2 + (k & 1)] = (_Float16)v[q];
      }
    }
  } else {
    for (int idx = tid; idx < Cin * 32; idx += nth) {
      int c = idx >> 5, j = idx & 31;
      int e = e0 + j;
      float v = (e < E) ? sb[(long)c * E + e] : 0.0f;
      lF[(j >> 4) * bank + (c >> 1) * 32 + (j & 15) * 2 + (c & 1)] = (_Float16)v;
    }
  }
  for (int idx = tid; idx < (Kpad - Kact) * 32; idx += nth) {
    int k = Kact + (idx >> 5), j = idx & 31;
    lF[(j >> 4) * bank + (k >> 1) * 32 + (j & 15) * 2 + (k & 1)] = (_Float16)0.0f;
  }
  __syncthreads();

  const int wave = tid >> 5, lane = tid & 31;
  const int orow0 = wave * 16;
  const int m  = lane & 15;        // A row / D column
  const int hi = lane >> 4;        // lane half
  const int kA = hi * 8;           // A K-offset per ISA 16-bit A layout
  const int kB = hi * 16;          // unused; B reads use kA, same k order as A
  const _Float16* wrow = Wk + (long)(orow0 + m) * Kpad;

  v8f acc0 = {}, acc1 = {};
  for (int kk = 0; kk < Kpad; kk += 32) {
    v16h a, b0, b1;
#pragma unroll
    for (int p = 0; p < 8; p++) {
      int ka = kk + ((p & 4) << 2) + kA + ((p & 3) << 1);  // even
      a[2*p]   = wrow[ka];
      a[2*p+1] = wrow[ka + 1];
      int kb = kk + ((p & 4) << 2) + kA + ((p & 3) << 1);  // even
      const _Float16* lp = lF + (kb >> 1) * 32 + m * 2;
      b0[2*p]   = lp[0];
      b0[2*p+1] = lp[1];
      b1[2*p]   = lp[bank];
      b1[2*p+1] = lp[bank + 1];
    }
    acc0 = __builtin_amdgcn_wmma_f32_16x16x32_f16(
        false, a, false, b0, (short)0, acc0, false, false);
    acc1 = __builtin_amdgcn_wmma_f32_16x16x32_f16(
        false, a, false, b1, (short)0, acc1, false, false);
  }

  float* db = dst + (long)b * dstBS;
  const int mb = hi * 8;
  const bool t0ok = e0 < E, t1ok = (e0 + 16) < E;
#pragma unroll
  for (int v = 0; v < 8; v++) {
    int o = orow0 + mb + v;
    if (o < O) {
      float bv = bias ? bias[o] : 0.0f;
      if (t0ok) db[(long)o * E + e0 + m]      = acc0[v] + bv;
      if (t1ok) db[(long)o * E + e0 + 16 + m] = acc1[v] + bv;
    }
  }
}

// ---------------------------------------------------------------------------
// inorm stats: mean / rstd over E per (b, c)
// ---------------------------------------------------------------------------
__global__ __launch_bounds__(256) void stats_kernel(
    const float* __restrict__ src, long bstride, int C, int E,
    float* __restrict__ mean, float* __restrict__ rstd) {
  int c = blockIdx.x, b = blockIdx.y;
  const float* row = src + (long)b * bstride + (long)c * E;
  float s = 0.f, s2 = 0.f;
  for (int i = threadIdx.x; i < E; i += 256) { float v = row[i]; s += v; s2 += v * v; }
  __shared__ float sh0[256], sh1[256];
  sh0[threadIdx.x] = s; sh1[threadIdx.x] = s2; __syncthreads();
  for (int off = 128; off > 0; off >>= 1) {
    if (threadIdx.x < off) { sh0[threadIdx.x] += sh0[threadIdx.x+off]; sh1[threadIdx.x] += sh1[threadIdx.x+off]; }
    __syncthreads();
  }
  if (threadIdx.x == 0) {
    float m = sh0[0] / E;
    float v = sh1[0] / E - m * m;
    mean[b * C + c] = m;
    rstd[b * C + c] = rsqrtf(v + 1e-5f);
  }
}

// mode 0: leaky_relu(0.2)   mode 1: sigmoid
__global__ __launch_bounds__(256) void norm_act_kernel(
    float* __restrict__ data, long bstride, int C, int E,
    const float* __restrict__ mean, const float* __restrict__ rstd, int mode) {
  int b = blockIdx.y;
  long i = (long)blockIdx.x * 256 + threadIdx.x;
  if (i >= (long)C * E) return;
  int c = (int)(i / E);
  float* p = data + (long)b * bstride + i;
  float v = (*p - mean[b * C + c]) * rstd[b * C + c];
  if (mode == 0)      v = v > 0.f ? v : 0.2f * v;
  else if (mode == 1) v = 1.0f / (1.0f + expf(-v));
  *p = v;
}

// softmax over {wa,wb} then blend: out = fa*w0 + fb*(1-w0)
__global__ __launch_bounds__(256) void fuse_kernel(
    const float* __restrict__ fa, const float* __restrict__ fb,
    const float* __restrict__ wa, const float* __restrict__ wb,
    float* __restrict__ out, long n) {
  long i = (long)blockIdx.x * 256 + threadIdx.x;
  if (i >= n) return;
  float ea = expf(wa[i]), eb = expf(wb[i]);
  float w0 = ea / (ea + eb);
  out[i] = fa[i] * w0 + fb[i] * (1.0f - w0);
}

// pool + seg head: seg_pred, softmax, argmax one-hot -> new_x2[B][6][EC]
__global__ __launch_bounds__(256) void pool_kernel(
    const float* __restrict__ x, const int* __restrict__ pool_idx,
    const float* __restrict__ dgW, const float* __restrict__ dgb,
    float* __restrict__ seg1, float* __restrict__ seg2,
    float* __restrict__ nx2, int E, int EC) {
  int b = blockIdx.y;
  int ec = blockIdx.x * 256 + threadIdx.x;
  if (ec >= EC) return;
  int pe = pool_idx[b * EC + ec];
  const float* xb = x + (long)b * 8 * E;
  float p0 = xb[5l*E + pe], p1 = xb[6l*E + pe], p2 = xb[7l*E + pe];
  float sp[3];
#pragma unroll
  for (int o = 0; o < 3; o++)
    sp[o] = dgb[o] + dgW[o*3+0]*p0 + dgW[o*3+1]*p1 + dgW[o*3+2]*p2;
  float mx = fmaxf(sp[0], fmaxf(sp[1], sp[2]));
  float ex[3]; float sum = 0.f;
#pragma unroll
  for (int o = 0; o < 3; o++) { ex[o] = expf(sp[o] - mx); sum += ex[o]; }
  int pm = 0;
  if (sp[1] > sp[pm]) pm = 1;
  if (sp[2] > sp[pm]) pm = 2;
  long base = (long)b * 3 * EC + ec;
  long nb   = (long)b * 6 * EC + ec;
#pragma unroll
  for (int o = 0; o < 3; o++) {
    seg1[base + (long)o * EC] = sp[o];
    seg2[base + (long)o * EC] = sp[o];
    nx2[nb + (long)o * EC]       = ex[o] / sum;
    nx2[nb + (long)(3 + o) * EC] = (pm == o) ? 1.0f : 0.0f;
  }
}

// unpool: fe_low / rounded / coarse outputs + feat_d/feat_e rows 19..21
__global__ __launch_bounds__(256) void unpool_kernel(
    const float* __restrict__ nx2, const int* __restrict__ unpool_idx,
    float* __restrict__ o_felow, float* __restrict__ o_round, float* __restrict__ o_coarse,
    float* __restrict__ FD, float* __restrict__ FE, int E, int EC) {
  int b = blockIdx.y;
  int e = blockIdx.x * 256 + threadIdx.x;
  if (e >= E) return;
  int u = unpool_idx[b * E + e];
  const float* nb = nx2 + (long)b * 6 * EC;
#pragma unroll
  for (int c = 0; c < 3; c++) {
    float fl = nb[(long)(3 + c) * EC + u];
    float r  = rintf(fl > 0.f ? 1.0f : fl);
    float co = rintf(fl);
    long oo = (long)b * 3 * E + (long)c * E + e;
    o_felow[oo] = fl; o_round[oo] = r; o_coarse[oo] = co;
    long fo = (long)b * 22 * E + (long)(19 + c) * E + e;
    FD[fo] = r; FE[fo] = co;
  }
}

// counts MLP -> counts_feature[B][3]
__global__ __launch_bounds__(256) void counts_kernel(
    const int* __restrict__ label,
    const float* __restrict__ W1, const float* __restrict__ b1,
    const float* __restrict__ W2, const float* __restrict__ b2,
    float* __restrict__ CF, int E) {
  int b = blockIdx.x;
  int cnt = 0;
  for (int i = threadIdx.x; i < E; i += 256) cnt += (label[b * E + i] != -1);
  __shared__ int sh[256];
  sh[threadIdx.x] = cnt; __syncthreads();
  for (int off = 128; off > 0; off >>= 1) {
    if (threadIdx.x < off) sh[threadIdx.x] += sh[threadIdx.x + off];
    __syncthreads();
  }
  if (threadIdx.x == 0) {
    float c = ((float)sh[0] - 3000.0f) / 19000.0f;
    float h[8];
#pragma unroll
    for (int i = 0; i < 8; i++) h[i] = c * W1[i] + b1[i];
#pragma unroll
    for (int o = 0; o < 3; o++) {
      float s = b2[o];
#pragma unroll
      for (int i = 0; i < 8; i++) s += W2[o * 8 + i] * h[i];
      CF[b * 3 + o] = s > 0.f ? s : (expf(s) - 1.0f);   // ELU
    }
  }
}

// getf conv (16x5) + counts broadcast -> feat_d / feat_e rows 0..18
__global__ __launch_bounds__(256) void getf_kernel(
    const float* __restrict__ x, const float* __restrict__ gW, const float* __restrict__ gb,
    const float* __restrict__ CF, float* __restrict__ FD, float* __restrict__ FE, int E) {
  int b = blockIdx.y;
  int e = blockIdx.x * 256 + threadIdx.x;
  if (e >= E) return;
  const float* xb = x + (long)b * 8 * E;
  float xi[5];
#pragma unroll
  for (int j = 0; j < 5; j++) xi[j] = xb[(long)j * E + e];
  long fb = (long)b * 22 * E + e;
#pragma unroll
  for (int o = 0; o < 16; o++) {
    float s = gb[o];
#pragma unroll
    for (int j = 0; j < 5; j++) s += gW[o * 5 + j] * xi[j];
    FD[fb + (long)o * E] = s; FE[fb + (long)o * E] = s;
  }
#pragma unroll
  for (int c = 0; c < 3; c++) {
    float v = CF[b * 3 + c];
    FD[fb + (long)(16 + c) * E] = v; FE[fb + (long)(16 + c) * E] = v;
  }
}

// ---------------------------------------------------------------------------
extern "C" void kernel_launch(void* const* d_in, const int* in_sizes, int n_in,
                              void* d_out, int out_size, void* d_ws, size_t ws_size,
                              hipStream_t stream) {
  const int E = E_FINE, EC = E_COARSE, B = NBATCH;
  const float* X      = (const float*)d_in[0];
  const int*   LABEL  = (const int*)  d_in[1];
  const int*   GEMMI  = (const int*)  d_in[2];
  const int*   POOL   = (const int*)  d_in[3];
  const int*   UNPOOL = (const int*)  d_in[4];
  int pi = 5;
  auto P = [&]() { return (const float*)d_in[pi++]; };
  const float *getf_W=P(), *getf_b=P(), *cnt_W1=P(), *cnt_b1=P(), *cnt_W2=P(), *cnt_b2=P();
  const float *c1_W=P(),  *c1_b=P(),  *c1c_W=P(), *c1c_b=P();
  const float *c2_W=P(),  *c2_b=P(),  *c2c_W=P(), *c2c_b=P();
  const float *c3_W=P(),  *c3_b=P(),  *c3c_W=P(), *c3c_b=P();
  const float *outd_W=P(), *outd_b=P(), *oute_W=P(), *oute_b=P();
  const float *at_a_W=P(),  *at_a_b=P(),  *at_b_W=P(),  *at_b_b=P();
  const float *at_al_W=P(), *at_al_b=P(), *at_bl_W=P(), *at_bl_b=P();
  const float *at_at_W=P(), *at_at_b=P(), *at_bt_W=P(), *at_bt_b=P();
  const float *at_af_W=P(), *at_af_b=P(), *at_bf_W=P(), *at_bf_b=P();
  const float *ca_W=P(), *cb_W=P(), *out_W=P(), *out_b=P(), *dg_W=P(), *dg_b=P();

  // output sections (tuple concatenated flat, f32)
  float* OUT      = (float*)d_out;
  float* o_feout  = OUT;
  float* o_seg1   = OUT + 264000;
  float* o_seg2   = OUT + 300000;
  float* o_felow  = OUT + 336000;
  float* o_round  = OUT + 600000;
  float* o_coarse = OUT + 864000;
  float* o_dout   = OUT + 1128000;
  float* o_eout   = OUT + 1392000;

  // workspace layout (f32)
  float* ws = (float*)d_ws;
  size_t off = 0;
  auto alloc = [&](size_t n) { float* p = ws + off; off += n; return p; };
  const long SL = (long)160 * E;        // slab batch stride (channels*E)
  float* S1 = alloc((size_t)SL * B);    // fea_d
  float* S2 = alloc((size_t)SL * B);    // fea_e
  float* S3 = alloc((size_t)SL * B);    // xa2 -> fuse
  float* S4 = alloc((size_t)SL * B);    // xb2 -> ca
  float* S5 = alloc((size_t)SL * B);    // wa  -> cb
  float* S6 = alloc((size_t)SL * B);    // x_all -> wb
  const long FL = (long)22 * E;
  float* FD  = alloc((size_t)FL * B);   // feat_d
  float* FE  = alloc((size_t)FL * B);   // feat_e
  float* NX2 = alloc((size_t)6 * EC * B);
  float* CF  = alloc(3 * B);
  float* MEAN = alloc(160 * B);
  float* RSTD = alloc(160 * B);
  _Float16* wArena = (_Float16*)(ws + off);
  size_t wOff = 0;

  // weight prep helper: f32 -> padded f16 [Opad][Kpad]
  auto prep = [&](const float* W, int O, int CinK, int Kpad, int Opad) {
    _Float16* dst = wArena + wOff; wOff += (size_t)Opad * Kpad;
    long n = (long)Opad * Kpad;
    wprep_kernel<<<dim3((n + 255) / 256), dim3(256), 0, stream>>>(W, dst, O, CinK, Kpad, Opad);
    return (const _Float16*)dst;
  };
  const _Float16* Wc1   = prep(c1_W,   32, 110, 128,  32);
  const _Float16* Wc1c  = prep(c1c_W,  32, 110, 128,  32);
  const _Float16* Wc2   = prep(c2_W,   64, 160, 160,  64);
  const _Float16* Wc2c  = prep(c2c_W,  64, 160, 160,  64);
  const _Float16* Wc3   = prep(c3_W,   64, 320, 320,  64);
  const _Float16* Wc3c  = prep(c3c_W,  64, 320, 320,  64);
  const _Float16* Wata  = prep(at_a_W, 40, 160, 160,  48);
  const _Float16* Watb  = prep(at_b_W, 40, 160, 160,  48);
  const _Float16* Watal = prep(at_al_W,80,  80,  96,  80);
  const _Float16* Watbl = prep(at_bl_W,80,  80,  96,  80);
  const _Float16* Watat = prep(at_at_W,80, 400, 416,  80);
  const _Float16* Watbt = prep(at_bt_W,80, 400, 416,  80);
  const _Float16* Wataf = prep(at_af_W,160,160, 160, 160);
  const _Float16* Watbf = prep(at_bf_W,160,160, 160, 160);
  const _Float16* Wca   = prep(ca_W,   64, 160, 160,  64);
  const _Float16* Wcb   = prep(cb_W,   32,  64,  64,  32);
  const _Float16* Woutd = prep(outd_W,  3, 160, 160,  16);
  const _Float16* Woute = prep(oute_W,  3, 160, 160,  16);
  const _Float16* Wout  = prep(out_W,   3,  32,  32,  16);

  // WMMA GEMM helper (32-edge tiles)
  auto gemm = [&](const float* src, long sBS, int Cin, const int* gi,
                  const _Float16* Wk, const float* bias,
                  float* dst, long dBS, int O, int Kpad, int Opad) {
    dim3 grid((E + 31) / 32, B);
    int threads = (Opad / 16) * 32;
    size_t shmem = (size_t)Kpad * 32 * sizeof(_Float16);
    wmma_gemm_kernel<<<grid, dim3(threads), shmem, stream>>>(
        src, sBS, Cin, gi, Wk, bias, dst, dBS, O, Kpad, E);
  };
  auto inorm_act = [&](float* base, long bstride, int C, int mode) {
    stats_kernel<<<dim3(C, B), dim3(256), 0, stream>>>(base, bstride, C, E, MEAN, RSTD);
    long n = (long)C * E;
    norm_act_kernel<<<dim3((n + 255) / 256, B), dim3(256), 0, stream>>>(
        base, bstride, C, E, MEAN, RSTD, mode);
  };

  // ---- front end ----
  pool_kernel<<<dim3((EC + 255) / 256, B), dim3(256), 0, stream>>>(
      X, POOL, dg_W, dg_b, o_seg1, o_seg2, NX2, E, EC);
  unpool_kernel<<<dim3((E + 255) / 256, B), dim3(256), 0, stream>>>(
      NX2, UNPOOL, o_felow, o_round, o_coarse, FD, FE, E, EC);
  counts_kernel<<<dim3(B), dim3(256), 0, stream>>>(
      LABEL, cnt_W1, cnt_b1, cnt_W2, cnt_b2, CF, E);
  getf_kernel<<<dim3((E + 255) / 256, B), dim3(256), 0, stream>>>(
      X, getf_W, getf_b, CF, FD, FE, E);

  // ---- branch d : FD(22) -> S1 rows [0..31][32..95][96..159] ----
  gemm(FD, FL, 22, GEMMI, Wc1, c1_b, S1,          SL, 32, 128, 32);
  inorm_act(S1,          SL, 32, 0);
  gemm(S1,          SL, 32, GEMMI, Wc2, c2_b, S1 + 32l * E, SL, 64, 160, 64);
  inorm_act(S1 + 32l * E, SL, 64, 0);
  gemm(S1 + 32l * E, SL, 64, GEMMI, Wc3, c3_b, S1 + 96l * E, SL, 64, 320, 64);
  inorm_act(S1 + 96l * E, SL, 64, 0);

  // ---- branch e : FE(22) -> S2 ----
  gemm(FE, FL, 22, GEMMI, Wc1c, c1c_b, S2,          SL, 32, 128, 32);
  inorm_act(S2,          SL, 32, 0);
  gemm(S2,          SL, 32, GEMMI, Wc2c, c2c_b, S2 + 32l * E, SL, 64, 160, 64);
  inorm_act(S2 + 32l * E, SL, 64, 0);
  gemm(S2 + 32l * E, SL, 64, GEMMI, Wc3c, c3c_b, S2 + 96l * E, SL, 64, 320, 64);
  inorm_act(S2 + 96l * E, SL, 64, 0);

  // ---- d_out / e_out heads ----
  gemm(S1, SL, 160, nullptr, Woutd, outd_b, o_dout, 3l * E, 3, 160, 16);
  gemm(S2, SL, 160, nullptr, Woute, oute_b, o_eout, 3l * E, 3, 160, 16);

  // ---- attention fuse ----
  gemm(S1, SL, 160, nullptr, Wata, at_a_b, S6,           SL, 40, 160, 48); // xa -> x_all[0:40]
  gemm(S2, SL, 160, nullptr, Watb, at_b_b, S6 + 40l * E, SL, 40, 160, 48); // xb -> x_all[40:80]
  gemm(S6, SL, 80, nullptr, Watal, at_al_b, S3,           SL, 80,  96, 80); // a_local
  gemm(S6, SL, 80, GEMMI,   Watat, at_at_b, S3 + 80l * E, SL, 80, 416, 80); // a_tri
  gemm(S6, SL, 80, nullptr, Watbl, at_bl_b, S4,           SL, 80,  96, 80); // b_local
  gemm(S6, SL, 80, GEMMI,   Watbt, at_bt_b, S4 + 80l * E, SL, 80, 416, 80); // b_tri
  gemm(S3, SL, 160, nullptr, Wataf, at_af_b, S5, SL, 160, 160, 160);        // wa lin
  inorm_act(S5, SL, 160, 1);                                                // sigmoid
  gemm(S4, SL, 160, nullptr, Watbf, at_bf_b, S6, SL, 160, 160, 160);        // wb lin
  inorm_act(S6, SL, 160, 1);                                                // sigmoid
  {
    long n = (long)B * 160 * E;
    fuse_kernel<<<dim3((n + 255) / 256), dim3(256), 0, stream>>>(S1, S2, S5, S6, S3, n);
  }

  // ---- tail ----
  gemm(S3, SL, 160, nullptr, Wca, nullptr, S4, SL, 64, 160, 64);
  inorm_act(S4, SL, 64, 0);
  gemm(S4, SL, 64, nullptr, Wcb, nullptr, S5, SL, 32, 64, 32);
  inorm_act(S5, SL, 32, 0);
  gemm(S5, SL, 32, nullptr, Wout, out_b, o_feout, 3l * E, 3, 32, 16);
}
